// CustomRNN_44985487459063
// MI455X (gfx1250) — compile-verified, matrix operand order checked by reading, not run on hardware
//
#include <hip/hip_runtime.h>
#include <hip/hip_bf16.h>
#include <math.h>

// CDNA5 wave32 WMMA types (probe-verified signatures)
typedef __attribute__((ext_vector_type(16))) __bf16 v16bf;
typedef __attribute__((ext_vector_type(8)))  float  v8f;

#define B_  64
#define T_  512
#define D_  1024
#define U_  1024
#define BT_ (B_ * T_)
#define RWG 64   // blocks in the persistent recurrent kernel

// 16-bit WMMA operand: regs 0-3 hold K-pairs 0..7, regs 4-7 hold K 16..23
// (upper half-wave +8) -> two contiguous 16-byte runs per chunk.
union ABf { v16bf v; uint4 q[2]; unsigned int u[8]; };

__device__ __forceinline__ unsigned int f32_to_bf16_bits(float f) {
  unsigned int u = __float_as_uint(f);
  return (u + 0x7FFFu + ((u >> 16) & 1u)) >> 16;  // RNE truncation
}
__device__ __forceinline__ unsigned int pack_bf16(float lo, float hi) {
  return (f32_to_bf16_bits(hi) << 16) | (f32_to_bf16_bits(lo) & 0xFFFFu);
}
__device__ __forceinline__ float bf16_bits_to_f32(unsigned short s) {
  return __uint_as_float(((unsigned int)s) << 16);
}
__device__ __forceinline__ float fast_tanh(float x) {
#if __has_builtin(__builtin_amdgcn_tanhf)
  return __builtin_amdgcn_tanhf(x);        // gfx1250 v_tanh_f32 (1 trans op)
#elif __has_builtin(__builtin_amdgcn_tanh_f32)
  return __builtin_amdgcn_tanh_f32(x);
#else
  return tanhf(x);
#endif
}

// ---------------------------------------------------------------------------
// Elementwise X -> bf16 (packed pair stores, fully coalesced).
// ---------------------------------------------------------------------------
__global__ __launch_bounds__(256) void convert_x(
    const float2* __restrict__ X2, unsigned int* __restrict__ Xbu) {
  long i = (long)blockIdx.x * blockDim.x + threadIdx.x;
  if (i < (long)BT_ * D_ / 2) {
    float2 v = X2[i];
    Xbu[i] = pack_bf16(v.x, v.y);
  }
}

// ---------------------------------------------------------------------------
// W1, W2 -> *transposed* bf16: W1t[u*D + d] = bf16(W1[d][u]) so the WMMA
// B-operand K-runs are contiguous.
// ---------------------------------------------------------------------------
__global__ __launch_bounds__(256) void convert_weights(
    const float* __restrict__ W1, const float* __restrict__ W2,
    unsigned short* __restrict__ W1t, unsigned short* __restrict__ W2t) {
  int idx = blockIdx.x * blockDim.x + threadIdx.x;
  if (idx < U_ * D_) {
    int u = idx / D_, d = idx % D_;
    W1t[idx] = (unsigned short)f32_to_bf16_bits(W1[d * U_ + u]);
  } else {
    int j = idx - U_ * D_;
    if (j < U_ * U_) {
      int n = j / U_, k = j % U_;
      W2t[j] = (unsigned short)f32_to_bf16_bits(W2[k * U_ + n]);
    }
  }
}

// ---------------------------------------------------------------------------
// Per-launch init (graph-replay deterministic): state0 = 0, barrier counters
// = 0, out = bc.
// ---------------------------------------------------------------------------
__global__ __launch_bounds__(256) void init_state(
    unsigned short* __restrict__ Sb0, int* __restrict__ cnt,
    float* __restrict__ out, const float* __restrict__ bc) {
  int idx = blockIdx.x * blockDim.x + threadIdx.x;
  if (idx < B_ * U_) Sb0[idx] = 0;  // bf16 zero
  if (idx < T_)      cnt[idx] = 0;
  if (idx < BT_)     out[idx] = bc[0];
}

// ---------------------------------------------------------------------------
// H = tanh(X @ W1 + b1), stored bf16 as [T][B][U] (dense per-step tiles for
// the scan). Grid (2048, 2) x 256 threads; each wave computes a 16x64 strip:
// A chunk loaded once (2 x b128), reused across 4 B tiles -> 4 WMMAs.
// ---------------------------------------------------------------------------
__global__ __launch_bounds__(256) void proj_kernel(
    const unsigned short* __restrict__ Xb, const unsigned short* __restrict__ W1t,
    const float* __restrict__ b1, unsigned short* __restrict__ Hb) {
  const int lane   = threadIdx.x & 31;
  const int wave   = threadIdx.x >> 5;
  const int mtile  = blockIdx.x;                       // 0..2047 (BT/16)
  const int ntile0 = (blockIdx.y * 8 + wave) * 4;      // 0..60 step 4
  const int m  = mtile * 16 + (lane & 15);
  const int hi = (lane >> 4) << 3;                     // upper half-wave: +8

  const unsigned short* xrow = Xb + (long)m * D_;

  v8f acc[4] = {};
  for (int kb = 0; kb < D_; kb += 32) {
    ABf a;
    a.q[0] = *(const uint4*)(xrow + kb + hi);
    a.q[1] = *(const uint4*)(xrow + kb + hi + 16);
#pragma unroll
    for (int j = 0; j < 4; ++j) {
      const unsigned short* wrow =
          W1t + (long)((ntile0 + j) * 16 + (lane & 15)) * D_;
      ABf b;
      b.q[0] = *(const uint4*)(wrow + kb + hi);
      b.q[1] = *(const uint4*)(wrow + kb + hi + 16);
      acc[j] = __builtin_amdgcn_wmma_f32_16x16x32_bf16(
          false, a.v, false, b.v, (short)0, acc[j], false, false);
    }
  }

#pragma unroll
  for (int j = 0; j < 4; ++j) {
    const int n = (ntile0 + j) * 16 + (lane & 15);
    const float bias = b1[n];
#pragma unroll
    for (int r = 0; r < 8; ++r) {
      int row = mtile * 16 + r + hi;       // flat BT index
      int bidx = row >> 9, tidx = row & (T_ - 1);
      float h = fast_tanh(acc[j][r] + bias);
      Hb[((long)tidx * B_ + bidx) * U_ + n] =
          (unsigned short)f32_to_bf16_bits(h);
    }
  }
}

// ---------------------------------------------------------------------------
// Persistent scan: 512 steps, device-wide barrier per step. 64 blocks x 256
// threads (8 waves): block owns 16 columns (W2 tile in LDS, ds_load_b128),
// waves 0-3 / 4-7 split the K=1024 reduction in half for the same 4 row
// tiles -> sequential WMMA depth per step is 16, partials combined via LDS.
// Wc projection fused (width-16 shuffle reduce + f32 atomic).
// ---------------------------------------------------------------------------
__global__ __launch_bounds__(256) void recur_kernel(
    const unsigned short* __restrict__ Hb, const unsigned short* __restrict__ W2t,
    const float* __restrict__ b2, const float* __restrict__ Wc,
    unsigned short* __restrict__ Sb, int* __restrict__ cnt,
    float* __restrict__ out) {
  __shared__ unsigned short lds_w2[16 * U_];           // 32 KB
  __shared__ float lds_acc[4][8][32];                  // 4 KB partial sums

  const int lane  = threadIdx.x & 31;
  const int wave  = threadIdx.x >> 5;                  // 0..7
  const int wrow4 = wave & 3;                          // row tile 0..3
  const int kgrp  = wave >> 2;                         // K half 0/1
  const int ntile = blockIdx.x;                        // 0..63 -> column tile
  const int n  = ntile * 16 + (lane & 15);
  const int hi = (lane >> 4) << 3;
  const int mrow = wrow4 * 16 + (lane & 15);           // A-operand row

  // Stage this block's 16 W2t rows (contiguous 32 KB) into LDS once.
  {
    const unsigned int* src = (const unsigned int*)(W2t + (size_t)ntile * 16 * U_);
    unsigned int* dst = (unsigned int*)lds_w2;
    for (int i = threadIdx.x; i < 16 * U_ / 2; i += blockDim.x) dst[i] = src[i];
  }
  __syncthreads();

  const float bias = b2[n];
  const float wc   = Wc[n];
  const unsigned short* wl = lds_w2 + (lane & 15) * U_;
  const int kbase = kgrp * (U_ / 2);

  for (int t = 0; t < T_; ++t) {
    const unsigned short* Sin  = Sb + (size_t)(t & 1) * (B_ * U_);
    unsigned short*       Sout = Sb + (size_t)((t + 1) & 1) * (B_ * U_);
    const unsigned short* srow = Sin + (long)mrow * U_;

    v8f acc = {};
#pragma unroll 4
    for (int kk = 0; kk < U_ / 2; kk += 32) {
      const int kb = kbase + kk;
      ABf a, b;
      a.q[0] = *(const uint4*)(srow + kb + hi);
      a.q[1] = *(const uint4*)(srow + kb + hi + 16);
      b.q[0] = *(const uint4*)(wl + kb + hi);
      b.q[1] = *(const uint4*)(wl + kb + hi + 16);
      acc = __builtin_amdgcn_wmma_f32_16x16x32_bf16(
          false, a.v, false, b.v, (short)0, acc, false, false);
    }

    // combine the two K-halves through LDS
    if (kgrp == 1) {
#pragma unroll
      for (int r = 0; r < 8; ++r) lds_acc[wrow4][r][lane] = acc[r];
    }
    __syncthreads();

    if (kgrp == 0) {
      // prefetch next step's h tile while we finish this one
      if (t + 1 < T_)
        __builtin_prefetch(&Hb[((long)(t + 1) * B_ + wrow4 * 16 + (lane & 15)) * U_ + n], 0, 1);
#pragma unroll
      for (int r = 0; r < 8; ++r) {
        int row = wrow4 * 16 + r + hi;                 // batch index 0..63
        float s = acc[r] + lds_acc[wrow4][r][lane];
        float h = bf16_bits_to_f32(Hb[((long)t * B_ + row) * U_ + n]);
        float y = h + fast_tanh(s + bias);
        Sout[(long)row * U_ + n] = (unsigned short)f32_to_bf16_bits(y);
        float p = y * wc;                              // fused Wc projection
#pragma unroll
        for (int off = 8; off >= 1; off >>= 1) p += __shfl_xor(p, off, 16);
        if ((lane & 15) == 0) atomicAdd(&out[row * T_ + t], p);
      }
    }

    // device-wide barrier: all blocks finish step t before t+1 overwrites
    // the buffer read here. (Also fences lds_acc reuse.)
    __threadfence();
    __syncthreads();
    if (threadIdx.x == 0) {
      atomicAdd(&cnt[t], 1);
      while (__hip_atomic_load(&cnt[t], __ATOMIC_ACQUIRE,
                               __HIP_MEMORY_SCOPE_AGENT) < RWG) {
        __builtin_amdgcn_s_sleep(1);
      }
    }
    __syncthreads();
  }
}

// ---------------------------------------------------------------------------
extern "C" void kernel_launch(void* const* d_in, const int* in_sizes, int n_in,
                              void* d_out, int out_size, void* d_ws, size_t ws_size,
                              hipStream_t stream) {
  const float* X  = (const float*)d_in[0];
  const float* W1 = (const float*)d_in[1];
  const float* b1 = (const float*)d_in[2];
  const float* W2 = (const float*)d_in[3];
  const float* b2 = (const float*)d_in[4];
  const float* Wc = (const float*)d_in[5];
  const float* bc = (const float*)d_in[6];
  float* out = (float*)d_out;

  char* ws = (char*)d_ws;
  size_t off = 0;
  unsigned short* Xb  = (unsigned short*)(ws + off); off += (size_t)BT_ * D_ * 2;
  unsigned short* W1t = (unsigned short*)(ws + off); off += (size_t)U_ * D_ * 2;
  unsigned short* W2t = (unsigned short*)(ws + off); off += (size_t)U_ * U_ * 2;
  unsigned short* Hb  = (unsigned short*)(ws + off); off += (size_t)BT_ * U_ * 2;
  unsigned short* Sb  = (unsigned short*)(ws + off); off += (size_t)2 * B_ * U_ * 2;
  int*            cnt = (int*)(ws + off);            off += (size_t)T_ * 4;

  convert_x<<<(BT_ * (D_ / 2) + 255) / 256, 256, 0, stream>>>(
      (const float2*)X, (unsigned int*)Xb);
  convert_weights<<<(2 * U_ * D_ + 255) / 256, 256, 0, stream>>>(W1, W2, W1t, W2t);
  init_state<<<(B_ * U_ + 255) / 256, 256, 0, stream>>>(Sb, cnt, out, bc);
  proj_kernel<<<dim3(BT_ / 16, 2), 256, 0, stream>>>(Xb, W1t, b1, Hb);
  recur_kernel<<<RWG, 256, 0, stream>>>(Hb, W2t, b2, Wc, Sb, cnt, out);
}
